// TANet_44779329028392
// MI455X (gfx1250) — matrix operand order repaired and checked by reading, not compiled, not run on hardware
//
#include <hip/hip_runtime.h>
#include <hip/hip_bf16.h>
#include <cstdint>
#include <cstddef>

// ---------------------------------------------------------------------------
// Types
// ---------------------------------------------------------------------------
typedef _Float16 half_t;
typedef __attribute__((ext_vector_type(16))) _Float16 v16h;
typedef __attribute__((ext_vector_type(8)))  _Float16 v8h;
typedef __attribute__((ext_vector_type(8)))  float    v8f;

#define N_STEPS 50
#define STEPSZ  0.02f

__device__ __forceinline__ float lrelu_f(float v) { return v >= 0.f ? v : 0.2f * v; }

// ---------------------------------------------------------------------------
// vol (f32, 1 channel) -> channel-last f16 [V][16], ch0 = vol, ch1..15 = 0
// ---------------------------------------------------------------------------
__global__ void cast_vol_cl_kernel(const float* __restrict__ in,
                                   half_t* __restrict__ out, int V) {
    int i = blockIdx.x * blockDim.x + threadIdx.x;
    if (i >= V * 16) return;
    int c = i & 15, v = i >> 4;
    out[i] = (c == 0) ? (half_t)in[v] : (half_t)0.f;
}

// ---------------------------------------------------------------------------
// Weight conversion: real layout [Cout][CinReal][27] f32 -> tap-major padded
// [Cout][27][CINP] f16 where the GEMM channel axis matches the channel-last
// storage concat:  ci in [0,SA) -> tensor-A ch ci (pad if >= realA),
//                  ci in [SA,SA+SB) -> tensor-B ch ci-SA (pad if >= realB),
//                  else pad.  Pad entries are 0 so WMMA sees exact zeros.
// ---------------------------------------------------------------------------
__global__ void cvt_weights_tapmajor_kernel(const float* __restrict__ w,
                                            half_t* __restrict__ o, int Cout,
                                            int realA, int SA, int realB, int SB,
                                            int CINP) {
    int i = blockIdx.x * blockDim.x + threadIdx.x;
    if (i >= Cout * 27 * CINP) return;
    int ci = i % CINP;
    int t  = (i / CINP) % 27;
    int co = i / (27 * CINP);
    int cr = -1;
    if (ci < SA) {
        if (ci < realA) cr = ci;
    } else if (ci < SA + SB) {
        int cj = ci - SA;
        if (cj < realB) cr = realA + cj;
    }
    o[i] = (cr >= 0)
               ? (half_t)w[((size_t)co * (realA + realB) + cr) * 27 + t]
               : (half_t)0.f;
}

// ---------------------------------------------------------------------------
// Implicit-GEMM 3D conv (pad=1, kernel 3^3) on v_wmma_f32_16x16x32_f16.
//   Channel-last activations: inA = [Vi][SA], inB = [Vi][SB], out = [Vo][16*MT].
//   GEMM: M = Cout = 16*MT (B fragment reused across M tiles),
//         N = 16 output voxels / wave, K = 27*CINP swept tap-by-tap.
//   B fragment per K-block = 16 *contiguous* f16 per lane -> two b128 loads.
//   Fragment layouts used here (same contiguous K order for A and B):
//     A 16x32 f16 : lane L row M=L&15, half j -> K = j + (L>=16)*16
//     B 32x16 f16 : lane L col N=L&15, half j -> K = j + (L>=16)*16
//     C/D 16x16   : VGPR r -> M = r + (L>=16)*8, N = L&15
// ---------------------------------------------------------------------------
template <int CINP, int MT, int SA, int SB>
__global__ __launch_bounds__(128)
void conv3d_wmma_kernel(const half_t* __restrict__ inA,
                        const half_t* __restrict__ inB,
                        const half_t* __restrict__ w16,   // [16*MT][27][CINP]
                        const float*  __restrict__ bias,  // [16*MT]
                        half_t* __restrict__ out,         // [Vo][16*MT]
                        int Di, int Hi, int Wi,
                        int Do, int Ho, int Wo,
                        int stride) {
    const int Kw    = 27 * CINP;
    const int No    = Do * Ho * Wo;
    const int lane  = threadIdx.x & 31;
    const int wid   = threadIdx.x >> 5;
    const int nBase = (blockIdx.x * 4 + wid) * 16;
    if (nBase >= No) return;                 // wave-uniform: EXEC all-1s for WMMA
    const int grp    = lane >> 4;            // lane group 0/1
    const int lanelo = lane & 15;

    // This lane's output column (B/C/D column N = lane&15)
    const int  n      = nBase + lanelo;
    const bool nvalid = (n < No);
    int od = 0, oh = 0, ow = 0;
    if (nvalid) {
        od = n / (Ho * Wo);
        int r = n - od * Ho * Wo;
        oh = r / Wo;
        ow = r - oh * Wo;
    }
    const int idB = od * stride - 1, ihB = oh * stride - 1, iwB = ow * stride - 1;

    // A-fragment row pointers (A row M = lane&15 within each 16-row M tile)
    const half_t* wrow[MT];
#pragma unroll
    for (int mt = 0; mt < MT; ++mt)
        wrow[mt] = w16 + (size_t)(mt * 16 + lanelo) * Kw + grp * 16;

    v8f acc[MT];
#pragma unroll
    for (int mt = 0; mt < MT; ++mt) acc[mt] = {};

    const v16h bzero = {};
    int kd = 0, kh = 0, kw = 0;              // incremental tap decomposition
    for (int tap = 0; tap < 27; ++tap) {
        const int  id = idB + kd, ih = ihB + kh, iw = iwB + kw;
        const bool okTap = nvalid && (unsigned)id < (unsigned)Di &&
                           (unsigned)ih < (unsigned)Hi && (unsigned)iw < (unsigned)Wi;
        const size_t sp  = ((size_t)(id < 0 ? 0 : id) * Hi + (ih < 0 ? 0 : ih)) * Wi +
                           (iw < 0 ? 0 : iw);
        const int kTap = tap * CINP;

        // Prefetch next tap's weight rows (streaming) -> global_prefetch_b8
        if (tap < 26) __builtin_prefetch(wrow[0] + kTap + CINP, 0, 3);

#pragma unroll
        for (int cb = 0; cb < CINP; cb += 32) {
            const int kBlk = kTap + cb;
            // ---- B fragment: this lane's 16 halves are channels
            //      [rs, rs+16) at voxel sp -> contiguous -> two b128 loads.
            const int rs = cb + grp * 16;    // run start (multiple of 16)
            v16h b;
            const bool inRange = rs < (SA + SB);
            const half_t* p = (rs < SA) ? (inA + sp * (size_t)SA + rs)
                                        : (inB + sp * (size_t)SB + (rs - SA));
            if (okTap && inRange) {
                v8h lo = *(const v8h*)(p);
                v8h hi = *(const v8h*)(p + 8);
                b = __builtin_shufflevector(lo, hi, 0, 1, 2, 3, 4, 5, 6, 7, 8, 9,
                                            10, 11, 12, 13, 14, 15);
            } else {
                b = bzero;
            }
            // ---- A fragments + WMMA per M tile (B reused across tiles)
#pragma unroll
            for (int mt = 0; mt < MT; ++mt) {
                v16h a;
#pragma unroll
                for (int j = 0; j < 8; ++j) {
                    a[j]     = wrow[mt][kBlk + j];
                    a[j + 8] = wrow[mt][kBlk + 8 + j];
                }
                acc[mt] = __builtin_amdgcn_wmma_f32_16x16x32_f16(
                    false, a, false, b, (short)0, acc[mt], false, false);
            }
        }
        if (++kw == 3) { kw = 0; if (++kh == 3) { kh = 0; ++kd; } }
    }

    // ---- Epilogue: bias + LeakyReLU; lane owns 8 contiguous output channels
    //      per M tile -> one b128 store each (channel-last output).
    if (nvalid) {
#pragma unroll
        for (int mt = 0; mt < MT; ++mt) {
            const int co0 = mt * 16 + grp * 8;
            v8h st;
#pragma unroll
            for (int r = 0; r < 8; ++r)
                st[r] = (half_t)lrelu_f(acc[mt][r] + bias[co0 + r]);
            *(v8h*)(out + (size_t)n * (16 * MT) + co0) = st;
        }
    }
}

// ---------------------------------------------------------------------------
// Small scalar conv (flow heads, Cout = 6, stride 1):
//   channel-last f16 input [V][SC] -> channel-major f32 output [Cout][V].
// ---------------------------------------------------------------------------
__global__ __launch_bounds__(256)
void conv3d_small_kernel(const half_t* __restrict__ in,   // [D*H*W][SC]
                         const float*  __restrict__ w,    // [Cout][Cin][27]
                         const float*  __restrict__ bias, // [Cout]
                         float* __restrict__ out,         // [Cout][D*H*W]
                         int Cin, int SC, int Cout, int D, int H, int W) {
    int Np  = D * H * W;
    int idx = blockIdx.x * blockDim.x + threadIdx.x;
    if (idx >= Cout * Np) return;
    int co = idx / Np, n = idx - co * Np;
    int d = n / (H * W);
    int r = n - d * H * W;
    int h = r / W, wv = r - (r / W) * W;
    float        s  = bias[co];
    const float* wr = w + (size_t)co * Cin * 27;
#pragma unroll
    for (int kd = 0; kd < 3; ++kd) {
        int id = d + kd - 1;
        if ((unsigned)id >= (unsigned)D) continue;
#pragma unroll
        for (int kh = 0; kh < 3; ++kh) {
            int ih = h + kh - 1;
            if ((unsigned)ih >= (unsigned)H) continue;
#pragma unroll
            for (int kw = 0; kw < 3; ++kw) {
                int iw = wv + kw - 1;
                if ((unsigned)iw >= (unsigned)W) continue;
                const half_t* ip = in + ((size_t)(id * H + ih) * W + iw) * SC;
                for (int ci = 0; ci < Cin; ++ci)
                    s += wr[ci * 27 + kd * 9 + kh * 3 + kw] * (float)ip[ci];
            }
        }
    }
    out[idx] = s;
}

// ---------------------------------------------------------------------------
// 2x trilinear upsample (matches up_axis: pos = o*(d-1)/(2d-1), clamp)
// ---------------------------------------------------------------------------
__device__ __forceinline__ void up_coord(int o, int d, int& i0, float& f) {
    float pos = (float)o * ((float)(d - 1) / (float)(2 * d - 1));
    int   i   = (int)floorf(pos);
    if (i > d - 2) i = d - 2;
    if (i < 0) i = 0;
    i0 = i;
    f  = pos - (float)i;
}

// Channel-last f16 variant: in [Vi][C], out [Vo][C] (C compile-time).
template <int C>
__global__ __launch_bounds__(256)
void upsample2x_cl_kernel(const half_t* __restrict__ in, half_t* __restrict__ out,
                          int Di, int Hi, int Wi) {
    int    Do = 2 * Di, Ho = 2 * Hi, Wo = 2 * Wi;
    size_t No  = (size_t)Do * Ho * Wo;
    size_t idx = (size_t)blockIdx.x * blockDim.x + threadIdx.x;
    if (idx >= No * C) return;
    int    c = (int)(idx % C);
    size_t n = idx / C;
    int    od = (int)(n / ((size_t)Ho * Wo));
    int    r  = (int)(n - (size_t)od * Ho * Wo);
    int    oh = r / Wo, ow = r - (r / Wo) * Wo;
    int   d0, h0, w0;
    float fd, fh, fw;
    up_coord(od, Di, d0, fd);
    up_coord(oh, Hi, h0, fh);
    up_coord(ow, Wi, w0, fw);
    auto at = [&](int d, int h, int w) -> float {
        return (float)in[((size_t)(d * Hi + h) * Wi + w) * C + c];
    };
    float v =
        (1 - fd) * ((1 - fh) * ((1 - fw) * at(d0, h0, w0) + fw * at(d0, h0, w0 + 1)) +
                    fh * ((1 - fw) * at(d0, h0 + 1, w0) + fw * at(d0, h0 + 1, w0 + 1))) +
        fd * ((1 - fh) * ((1 - fw) * at(d0 + 1, h0, w0) + fw * at(d0 + 1, h0, w0 + 1)) +
              fh * ((1 - fw) * at(d0 + 1, h0 + 1, w0) + fw * at(d0 + 1, h0 + 1, w0 + 1)));
    out[idx] = (half_t)v;
}

// Channel-major f32 variant (flow stacks, C = 6).
__global__ __launch_bounds__(256)
void upsample2x_cm_kernel(const float* __restrict__ in, float* __restrict__ out,
                          int C, int Di, int Hi, int Wi) {
    int    Do = 2 * Di, Ho = 2 * Hi, Wo = 2 * Wi;
    size_t No  = (size_t)Do * Ho * Wo;
    size_t idx = (size_t)blockIdx.x * blockDim.x + threadIdx.x;
    if (idx >= (size_t)C * No) return;
    int    c = (int)(idx / No);
    size_t n = idx - (size_t)c * No;
    int    od = (int)(n / ((size_t)Ho * Wo));
    int    r  = (int)(n - (size_t)od * Ho * Wo);
    int    oh = r / Wo, ow = r - (r / Wo) * Wo;
    int   d0, h0, w0;
    float fd, fh, fw;
    up_coord(od, Di, d0, fd);
    up_coord(oh, Hi, h0, fh);
    up_coord(ow, Wi, w0, fw);
    const float* base = in + (size_t)c * Di * Hi * Wi;
    auto at = [&](int d, int h, int w) -> float {
        return base[((size_t)d * Hi + h) * Wi + w];
    };
    float v =
        (1 - fd) * ((1 - fh) * ((1 - fw) * at(d0, h0, w0) + fw * at(d0, h0, w0 + 1)) +
                    fh * ((1 - fw) * at(d0, h0 + 1, w0) + fw * at(d0, h0 + 1, w0 + 1))) +
        fd * ((1 - fh) * ((1 - fw) * at(d0 + 1, h0, w0) + fw * at(d0 + 1, h0, w0 + 1)) +
              fh * ((1 - fw) * at(d0 + 1, h0 + 1, w0) + fw * at(d0 + 1, h0 + 1, w0 + 1)));
    out[idx] = v;
}

// ---------------------------------------------------------------------------
// Pack [6][V] x3 flow stacks -> channel-last SVF table [V][18] (72 B / voxel)
// ---------------------------------------------------------------------------
__global__ __launch_bounds__(256)
void pack_svf_kernel(const float* __restrict__ vf1, const float* __restrict__ vf2,
                     const float* __restrict__ vf3, float* __restrict__ svf, int V) {
    int idx = blockIdx.x * blockDim.x + threadIdx.x;
    if (idx >= V * 18) return;
    int v = idx / 18, g = idx - v * 18;
    const float* src = (g < 6) ? vf1 : (g < 12 ? vf2 : vf3);
    int          gc  = (g < 6) ? g : (g < 12 ? g - 6 : g - 12);
    svf[idx] = src[(size_t)gc * V + v];
}

// ---------------------------------------------------------------------------
// att_net: 50x1 -> 64 -> 128 -> 128 -> 64 -> 6 (+softmax). One workgroup.
// ---------------------------------------------------------------------------
__global__ __launch_bounds__(128)
void att_net_kernel(const float* __restrict__ w1, const float* __restrict__ b1,
                    const float* __restrict__ w2, const float* __restrict__ b2,
                    const float* __restrict__ w3, const float* __restrict__ b3,
                    const float* __restrict__ w4, const float* __restrict__ b4,
                    const float* __restrict__ w5, const float* __restrict__ b5,
                    float* __restrict__ wts /* [50][6] */) {
    __shared__ float A[N_STEPS * 128];
    __shared__ float B[N_STEPS * 128];
    int tid = threadIdx.x;
    for (int i = tid; i < N_STEPS * 64; i += 128) {           // fc1
        int t = i / 64, j = i - t * 64;
        A[t * 64 + j] = lrelu_f(((float)t * STEPSZ) * w1[j] + b1[j]);
    }
    __syncthreads();
    for (int i = tid; i < N_STEPS * 128; i += 128) {          // fc2
        int t = i / 128, j = i - t * 128;
        float s = b2[j];
        for (int c = 0; c < 64; ++c) s += A[t * 64 + c] * w2[j * 64 + c];
        B[t * 128 + j] = lrelu_f(s);
    }
    __syncthreads();
    for (int i = tid; i < N_STEPS * 128; i += 128) {          // fc3
        int t = i / 128, j = i - t * 128;
        float s = b3[j];
        for (int c = 0; c < 128; ++c) s += B[t * 128 + c] * w3[j * 128 + c];
        A[t * 128 + j] = lrelu_f(s);
    }
    __syncthreads();
    for (int i = tid; i < N_STEPS * 64; i += 128) {           // fc4
        int t = i / 64, j = i - t * 64;
        float s = b4[j];
        for (int c = 0; c < 128; ++c) s += A[t * 128 + c] * w4[j * 128 + c];
        B[t * 64 + j] = lrelu_f(s);
    }
    __syncthreads();
    if (tid < N_STEPS) {                                       // fc5 + softmax
        float lg[6], mx = -1e30f;
#pragma unroll
        for (int j = 0; j < 6; ++j) {
            float s = b5[j];
            for (int c = 0; c < 64; ++c) s += B[tid * 64 + c] * w5[j * 64 + c];
            lg[j] = s;
            mx    = fmaxf(mx, s);
        }
        float den = 0.f;
#pragma unroll
        for (int j = 0; j < 6; ++j) { lg[j] = __expf(lg[j] - mx); den += lg[j]; }
#pragma unroll
        for (int j = 0; j < 6; ++j) wts[tid * 6 + j] = lg[j] / den;
    }
}

// ---------------------------------------------------------------------------
// ODE integrator: 50 steps per point in registers; channel-last SVF gathers
// (8 corners x 72 B) served from the L2-resident 35 MB table.
// ---------------------------------------------------------------------------
__global__ __launch_bounds__(256)
void integrate_kernel(const float* __restrict__ svf /* [D][H][W][18] */,
                      const float* __restrict__ wts /* [50][6] */,
                      const float* __restrict__ x_in, float* __restrict__ x_out,
                      int nV) {
    __shared__ float sw[N_STEPS * 6];
    for (int i = threadIdx.x; i < N_STEPS * 6; i += blockDim.x) sw[i] = wts[i];
    __syncthreads();
    int v = blockIdx.x * blockDim.x + threadIdx.x;
    if (v >= nV) return;
    const int D = 64, H = 96, W = 80;
    float pd = x_in[v * 3 + 0], ph = x_in[v * 3 + 1], pw = x_in[v * 3 + 2];
    for (int t = 0; t < N_STEPS; ++t) {
        float cd = fminf(fmaxf(pd, 0.f), (float)(D - 1));
        float ch = fminf(fmaxf(ph, 0.f), (float)(H - 1));
        float cw = fminf(fmaxf(pw, 0.f), (float)(W - 1));
        int d0 = (int)floorf(cd); d0 = d0 < 0 ? 0 : (d0 > D - 2 ? D - 2 : d0);
        int h0 = (int)floorf(ch); h0 = h0 < 0 ? 0 : (h0 > H - 2 ? H - 2 : h0);
        int w0 = (int)floorf(cw); w0 = w0 < 0 ? 0 : (w0 > W - 2 ? W - 2 : w0);
        float fd = cd - (float)d0, fh = ch - (float)h0, fw = cw - (float)w0;
        float acc[18];
#pragma unroll
        for (int s = 0; s < 18; ++s) acc[s] = 0.f;
#pragma unroll
        for (int dd = 0; dd < 2; ++dd)
#pragma unroll
            for (int hh = 0; hh < 2; ++hh)
#pragma unroll
                for (int ww = 0; ww < 2; ++ww) {
                    float cwt = (dd ? fd : 1.f - fd) * (hh ? fh : 1.f - fh) *
                                (ww ? fw : 1.f - fw);
                    const float* p =
                        svf + ((size_t)((d0 + dd) * H + (h0 + hh)) * W + (w0 + ww)) * 18;
#pragma unroll
                    for (int s = 0; s < 18; ++s) acc[s] += cwt * p[s];
                }
        const float* wt = sw + t * 6;
        float vx = 0.f, vy = 0.f, vz = 0.f;
#pragma unroll
        for (int s = 0; s < 6; ++s) {
            vx += wt[s] * acc[s * 3 + 0];
            vy += wt[s] * acc[s * 3 + 1];
            vz += wt[s] * acc[s * 3 + 2];
        }
        pd += STEPSZ * vx;
        ph += STEPSZ * vy;
        pw += STEPSZ * vz;
    }
    x_out[v * 3 + 0] = pd;
    x_out[v * 3 + 1] = ph;
    x_out[v * 3 + 2] = pw;
}

// ---------------------------------------------------------------------------
// Host orchestration
// ---------------------------------------------------------------------------
template <int CINP, int MT, int SA, int SB>
static void launch_conv(const half_t* iA, const half_t* iB, const half_t* w16,
                        const float* bias, half_t* o,
                        int Di, int Hi, int Wi, int Do, int Ho, int Wo, int st,
                        hipStream_t stream) {
    int  No = Do * Ho * Wo;
    dim3 grid((No + 63) / 64);
    conv3d_wmma_kernel<CINP, MT, SA, SB><<<grid, 128, 0, stream>>>(
        iA, iB, w16, bias, o, Di, Hi, Wi, Do, Ho, Wo, st);
}

extern "C" void kernel_launch(void* const* d_in, const int* in_sizes, int n_in,
                              void* d_out, int out_size, void* d_ws, size_t ws_size,
                              hipStream_t stream) {
    (void)in_sizes; (void)n_in; (void)out_size; (void)ws_size;
    const int D0 = 64, H0 = 96, W0 = 80;  const int V0 = D0 * H0 * W0;
    const int D1 = 32, H1 = 48, W1 = 40;  const int V1 = D1 * H1 * W1;
    const int D2 = 16, H2 = 24, W2 = 20;  const int V2 = D2 * H2 * W2;
    const int D3 = 8,  H3 = 12, W3 = 10;  const int V3 = D3 * H3 * W3;
    const int NV = 120000;

    // Inputs: setup_inputs() dict order, nested dicts flattened (w then b)
    const float* x_in = (const float*)d_in[0];
    const float* vol  = (const float*)d_in[1];
    const float *w_c1 = (const float*)d_in[2],  *b_c1 = (const float*)d_in[3];
    const float *w_c2 = (const float*)d_in[4],  *b_c2 = (const float*)d_in[5];
    const float *w_c3 = (const float*)d_in[6],  *b_c3 = (const float*)d_in[7];
    const float *w_c4 = (const float*)d_in[8],  *b_c4 = (const float*)d_in[9];
    const float *w_c5 = (const float*)d_in[10], *b_c5 = (const float*)d_in[11];
    const float *w_d4 = (const float*)d_in[12], *b_d4 = (const float*)d_in[13];
    const float *w_d3 = (const float*)d_in[14], *b_d3 = (const float*)d_in[15];
    const float *w_d2 = (const float*)d_in[16], *b_d2 = (const float*)d_in[17];
    const float *w_d1 = (const float*)d_in[18], *b_d1 = (const float*)d_in[19];
    const float *w_f1 = (const float*)d_in[20], *b_f1 = (const float*)d_in[21];
    const float *w_f2 = (const float*)d_in[22], *b_f2 = (const float*)d_in[23];
    const float *w_f3 = (const float*)d_in[24], *b_f3 = (const float*)d_in[25];
    const float *a_w1 = (const float*)d_in[26], *a_b1 = (const float*)d_in[27];
    const float *a_w2 = (const float*)d_in[28], *a_b2 = (const float*)d_in[29];
    const float *a_w3 = (const float*)d_in[30], *a_b3 = (const float*)d_in[31];
    const float *a_w4 = (const float*)d_in[32], *a_b4 = (const float*)d_in[33];
    const float *a_w5 = (const float*)d_in[34], *a_b5 = (const float*)d_in[35];

    // Bump allocator over d_ws
    char*  wsp = (char*)d_ws;
    size_t off = 0;
    auto alloc = [&](size_t bytes) -> void* {
        off = (off + 255) & ~(size_t)255;
        void* p = wsp + off;
        off += bytes;
        return p;
    };

    // Channel-last f16 activations + tap-major padded weights ([Cout][27][CINP])
    half_t* vol_h = (half_t*)alloc((size_t)V0 * 16 * 2);  // [V0][16], ch0 = vol
    half_t* wc1 = (half_t*)alloc((size_t)16 * 27 * 32 * 2);
    half_t* wc2 = (half_t*)alloc((size_t)32 * 27 * 32 * 2);
    half_t* wc3 = (half_t*)alloc((size_t)32 * 27 * 32 * 2);
    half_t* wc4 = (half_t*)alloc((size_t)32 * 27 * 32 * 2);
    half_t* wc5 = (half_t*)alloc((size_t)32 * 27 * 32 * 2);
    half_t* wd4 = (half_t*)alloc((size_t)32 * 27 * 64 * 2);
    half_t* wd3 = (half_t*)alloc((size_t)32 * 27 * 64 * 2);
    half_t* wd2 = (half_t*)alloc((size_t)32 * 27 * 64 * 2);
    half_t* wd1 = (half_t*)alloc((size_t)16 * 27 * 64 * 2);
    half_t* x1  = (half_t*)alloc((size_t)V0 * 16 * 2);    // [V0][16]
    half_t* x2  = (half_t*)alloc((size_t)V1 * 32 * 2);    // [V1][32]
    half_t* x3  = (half_t*)alloc((size_t)V2 * 32 * 2);
    half_t* x4  = (half_t*)alloc((size_t)V3 * 32 * 2);
    half_t* x5  = (half_t*)alloc((size_t)V3 * 32 * 2);
    half_t* d4  = (half_t*)alloc((size_t)V3 * 32 * 2);
    half_t* d4u = (half_t*)alloc((size_t)V2 * 32 * 2);
    half_t* d3  = (half_t*)alloc((size_t)V2 * 32 * 2);
    half_t* d3u = (half_t*)alloc((size_t)V1 * 32 * 2);
    half_t* d2  = (half_t*)alloc((size_t)V1 * 32 * 2);
    half_t* d2u = (half_t*)alloc((size_t)V0 * 32 * 2);
    half_t* d1  = (half_t*)alloc((size_t)V0 * 16 * 2);    // [V0][16]
    // f32 flow stacks (channel-major [6][V]) + packed SVF + attention weights
    float* vf1a = (float*)alloc((size_t)6 * V2 * 4);
    float* vf1b = (float*)alloc((size_t)6 * V1 * 4);
    float* vf1  = (float*)alloc((size_t)6 * V0 * 4);
    float* vf2a = (float*)alloc((size_t)6 * V1 * 4);
    float* vf2  = (float*)alloc((size_t)6 * V0 * 4);
    float* vf3  = (float*)alloc((size_t)6 * V0 * 4);
    float* svf  = (float*)alloc((size_t)18 * V0 * 4);
    float* wts  = (float*)alloc((size_t)N_STEPS * 6 * 4);

    auto cvtw = [&](const float* w, half_t* o, int Cout, int realA, int SA,
                    int realB, int SB, int CINP) {
        int n = Cout * 27 * CINP;
        cvt_weights_tapmajor_kernel<<<(n + 255) / 256, 256, 0, stream>>>(
            w, o, Cout, realA, SA, realB, SB, CINP);
    };

    // --- Prep: cast volume (padded channel-last), convert weights, att MLP
    cast_vol_cl_kernel<<<(V0 * 16 + 255) / 256, 256, 0, stream>>>(vol, vol_h, V0);
    cvtw(w_c1, wc1, 16, 1, 16, 0, 0, 32);
    cvtw(w_c2, wc2, 32, 16, 16, 0, 0, 32);
    cvtw(w_c3, wc3, 32, 32, 32, 0, 0, 32);
    cvtw(w_c4, wc4, 32, 32, 32, 0, 0, 32);
    cvtw(w_c5, wc5, 32, 32, 32, 0, 0, 32);
    cvtw(w_d4, wd4, 32, 32, 32, 32, 32, 64);
    cvtw(w_d3, wd3, 32, 32, 32, 32, 32, 64);
    cvtw(w_d2, wd2, 32, 32, 32, 32, 32, 64);
    cvtw(w_d1, wd1, 16, 32, 32, 16, 16, 64);
    att_net_kernel<<<1, 128, 0, stream>>>(a_w1, a_b1, a_w2, a_b2, a_w3, a_b3,
                                          a_w4, a_b4, a_w5, a_b5, wts);

    // --- Encoder (channel-last activations everywhere)
    launch_conv<32, 1, 16, 0>(vol_h, vol_h, wc1, b_c1, x1,
                              D0, H0, W0, D0, H0, W0, 1, stream);
    launch_conv<32, 2, 16, 0>(x1, x1, wc2, b_c2, x2,
                              D0, H0, W0, D1, H1, W1, 2, stream);
    launch_conv<32, 2, 32, 0>(x2, x2, wc3, b_c3, x3,
                              D1, H1, W1, D2, H2, W2, 2, stream);
    launch_conv<32, 2, 32, 0>(x3, x3, wc4, b_c4, x4,
                              D2, H2, W2, D3, H3, W3, 2, stream);
    launch_conv<32, 2, 32, 0>(x4, x4, wc5, b_c5, x5,
                              D3, H3, W3, D3, H3, W3, 1, stream);

    // --- Decoder with skip concats + upsampling
    launch_conv<64, 2, 32, 32>(x5, x4, wd4, b_d4, d4,
                               D3, H3, W3, D3, H3, W3, 1, stream);
    upsample2x_cl_kernel<32><<<((size_t)V2 * 32 + 255) / 256, 256, 0, stream>>>(
        d4, d4u, D3, H3, W3);
    launch_conv<64, 2, 32, 32>(d4u, x3, wd3, b_d3, d3,
                               D2, H2, W2, D2, H2, W2, 1, stream);

    conv3d_small_kernel<<<(6 * V2 + 255) / 256, 256, 0, stream>>>(
        d3, w_f1, b_f1, vf1a, 32, 32, 6, D2, H2, W2);
    upsample2x_cm_kernel<<<((size_t)6 * V1 + 255) / 256, 256, 0, stream>>>(
        vf1a, vf1b, 6, D2, H2, W2);
    upsample2x_cm_kernel<<<((size_t)6 * V0 + 255) / 256, 256, 0, stream>>>(
        vf1b, vf1, 6, D1, H1, W1);

    upsample2x_cl_kernel<32><<<((size_t)V1 * 32 + 255) / 256, 256, 0, stream>>>(
        d3, d3u, D2, H2, W2);
    launch_conv<64, 2, 32, 32>(d3u, x2, wd2, b_d2, d2,
                               D1, H1, W1, D1, H1, W1, 1, stream);

    conv3d_small_kernel<<<(6 * V1 + 255) / 256, 256, 0, stream>>>(
        d2, w_f2, b_f2, vf2a, 32, 32, 6, D1, H1, W1);
    upsample2x_cm_kernel<<<((size_t)6 * V0 + 255) / 256, 256, 0, stream>>>(
        vf2a, vf2, 6, D1, H1, W1);

    upsample2x_cl_kernel<32><<<((size_t)V0 * 32 + 255) / 256, 256, 0, stream>>>(
        d2, d2u, D1, H1, W1);
    launch_conv<64, 1, 32, 16>(d2u, x1, wd1, b_d1, d1,
                               D0, H0, W0, D0, H0, W0, 1, stream);

    conv3d_small_kernel<<<(6 * V0 + 255) / 256, 256, 0, stream>>>(
        d1, w_f3, b_f3, vf3, 16, 16, 6, D0, H0, W0);

    // --- Pack SVFs channel-last, then integrate 50 Euler steps per point
    pack_svf_kernel<<<(V0 * 18 + 255) / 256, 256, 0, stream>>>(vf1, vf2, vf3, svf, V0);
    integrate_kernel<<<(NV + 255) / 256, 256, 0, stream>>>(svf, wts, x_in,
                                                           (float*)d_out, NV);
}
